// GATv2_317827580334
// MI455X (gfx1250) — matrix operand order repaired and checked by reading, not compiled, not run on hardware
//
#include <hip/hip_runtime.h>

typedef __attribute__((ext_vector_type(16))) _Float16 v16h;
typedef __attribute__((ext_vector_type(8)))  _Float16 v8h;
typedef __attribute__((ext_vector_type(8)))  float    v8f;

#define NEG_SLOPE 0.2f
#define BN_EPS 1e-5f

// ---------------------------------------------------------------------------
// float atomic max via sign-aware integer atomics (works with -inf init)
// ---------------------------------------------------------------------------
__device__ __forceinline__ void atomicMaxF(float* addr, float val) {
    if (val >= 0.0f) {
        atomicMax((int*)addr, __float_as_int(val));
    } else {
        atomicMin((unsigned int*)addr, (unsigned int)__float_as_int(val));
    }
}

// ---------------------------------------------------------------------------
// small utility kernels
// ---------------------------------------------------------------------------
__global__ void k_cvt_f32_f16(const float* __restrict__ in, _Float16* __restrict__ out,
                              long long n) {
    long long t = blockIdx.x * (long long)blockDim.x + threadIdx.x;
    if (t < n) out[t] = (_Float16)in[t];
}

__global__ void k_fill(float* __restrict__ p, float val, long long n) {
    long long t = blockIdx.x * (long long)blockDim.x + threadIdx.x;
    if (t < n) p[t] = val;
}

// p[row*C + c] = bias[c]
__global__ void k_fill_rowbias(float* __restrict__ p, const float* __restrict__ bias,
                               int C, long long n) {
    long long t = blockIdx.x * (long long)blockDim.x + threadIdx.x;
    if (t < n) p[t] = bias[(int)(t % C)];
}

// Pack row-major W[K][Ncols] (f32) into f16 WMMA-B layout:
//   Bp[((k/32)*Ncols + col)*32 + (k%32)]
// so a lane reads its 16 B-halves (K = hi*16 .. hi*16+15 of the chunk) contiguously.
__global__ void k_packw(const float* __restrict__ W, _Float16* __restrict__ Bp,
                        int Ncols, long long n) {
    long long t = blockIdx.x * (long long)blockDim.x + threadIdx.x;
    if (t >= n) return;
    int k   = (int)(t / Ncols);
    int col = (int)(t % Ncols);
    int chunk = k >> 5, kl = k & 31;
    Bp[((size_t)chunk * Ncols + col) * 32 + kl] = (_Float16)W[t];
}

// ---------------------------------------------------------------------------
// WMMA GEMM: C[M x NCOLS] = A[M x K](f16,row-major) * Bp(packed f16) + bias
// One wave per 16x16 output tile; K-loop fully unrolled -> K/32 v_wmma ops.
// ---------------------------------------------------------------------------
template <int K, int NCOLS>
__global__ void k_gemm_wmma(const _Float16* __restrict__ A,
                            const _Float16* __restrict__ Bp,
                            const float* __restrict__ bias,
                            float* __restrict__ C, int Mtiles) {
    const int NT = NCOLS / 16;
    int wid = blockIdx.x * (blockDim.x >> 5) + (threadIdx.x >> 5);
    if (wid >= Mtiles * NT) return;   // uniform per wave: EXEC stays all-ones
    int lane = threadIdx.x & 31;
    int mt = wid / NT, nt = wid % NT;
    int l15 = lane & 15;
    int hi  = lane >> 4;              // half-wave select (K-halves / M-halves)
    int m = mt * 16 + l15;
    int n = nt * 16 + l15;

    const _Float16* arow = A + (size_t)m * K;
    v8f acc = {};

#pragma unroll
    for (int k0 = 0; k0 < K; k0 += 32) {
        // A, contiguous k order as used for B: lanes 0-15 hold K = k0+0..15, lanes 16-31 hold K = k0+16..31
        v8h alo = *(const v8h*)(arow + k0 + hi * 16);
        v8h ahi = *(const v8h*)(arow + k0 + hi * 16 + 8);
        v16h a;
#pragma unroll
        for (int i = 0; i < 8; ++i) { a[i] = alo[i]; a[8 + i] = ahi[i]; }
        // B: lanes 0-15 hold K=0..15 of chunk, lanes 16-31 hold K=16..31
        int chunk = k0 >> 5;
        v16h b = *(const v16h*)(Bp + ((size_t)chunk * NCOLS + n) * 32 + hi * 16);
        acc = __builtin_amdgcn_wmma_f32_16x16x32_f16(
            false, a, false, b, (short)0, acc, false, false);
    }
    // C/D layout: VGPR r -> row (r + 8*hi), col = lane&15
#pragma unroll
    for (int r = 0; r < 8; ++r) {
        int row = mt * 16 + hi * 8 + r;
        C[(size_t)row * NCOLS + n] = acc[r] + bias[n];
    }
}

// ---------------------------------------------------------------------------
// Edge kernels
// ---------------------------------------------------------------------------
// Layer-1 logits: one wave per edge, 8 heads x 32 channels.
__global__ void k_logits1(const float* __restrict__ H, const float* __restrict__ att,
                          const long long* __restrict__ ei, int E, int ET,
                          float* __restrict__ logits, float* __restrict__ mbuf) {
    int e = blockIdx.x * 8 + (threadIdx.x >> 5);
    int lane = threadIdx.x & 31;
    if (e >= ET) return;
    long long s, d;
    if (e < E) { s = ei[e]; d = ei[E + e]; } else { s = d = e - E; }
    const float* hs = H + (size_t)s * 256;
    const float* hd = H + (size_t)d * 256;
#pragma unroll
    for (int h = 0; h < 8; ++h) {
        int idx = h * 32 + lane;
        float v = hs[idx] + hd[idx];
        v = v > 0.f ? v : NEG_SLOPE * v;
        float acc = att[idx] * v;
#pragma unroll
        for (int off = 16; off > 0; off >>= 1) acc += __shfl_xor(acc, off, 32);
        if (lane == 0) {
            logits[(size_t)e * 8 + h] = acc;
            atomicMaxF(&mbuf[(size_t)d * 8 + h], acc);
        }
    }
}

// Layer-2 logits: one wave per edge, 1 head x 64 channels (2 per lane).
__global__ void k_logits2(const float* __restrict__ H, const float* __restrict__ att,
                          const long long* __restrict__ ei, int E, int ET,
                          float* __restrict__ logits, float* __restrict__ mbuf) {
    int e = blockIdx.x * 8 + (threadIdx.x >> 5);
    int lane = threadIdx.x & 31;
    if (e >= ET) return;
    long long s, d;
    if (e < E) { s = ei[e]; d = ei[E + e]; } else { s = d = e - E; }
    const float* hs = H + (size_t)s * 64;
    const float* hd = H + (size_t)d * 64;
    float acc = 0.f;
#pragma unroll
    for (int half = 0; half < 2; ++half) {
        int idx = lane + 32 * half;
        float v = hs[idx] + hd[idx];
        v = v > 0.f ? v : NEG_SLOPE * v;
        acc += att[idx] * v;
    }
#pragma unroll
    for (int off = 16; off > 0; off >>= 1) acc += __shfl_xor(acc, off, 32);
    if (lane == 0) {
        logits[e] = acc;
        atomicMaxF(&mbuf[d], acc);
    }
}

// exp(logit - max[dst]) in-place + atomic segment sum. One thread per (edge,head).
__global__ void k_expsum(float* __restrict__ logits, const float* __restrict__ mbuf,
                         float* __restrict__ sbuf, const long long* __restrict__ ei,
                         int E, int ET, int H) {
    long long t = blockIdx.x * (long long)blockDim.x + threadIdx.x;
    if (t >= (long long)ET * H) return;
    int e = (int)(t / H);
    int h = (int)(t % H);
    long long d = (e < E) ? ei[E + e] : (long long)(e - E);
    float ex = __expf(logits[t] - mbuf[d * H + h]);
    logits[t] = ex;
    atomicAdd(&sbuf[d * H + h], ex);
}

// alpha-weighted message scatter-add. One thread per (edge, channel).
__global__ void k_scatter(const float* __restrict__ H, const float* __restrict__ ex,
                          const float* __restrict__ sbuf, const long long* __restrict__ ei,
                          int E, int ET, int C, int Hh, int hshift,
                          float* __restrict__ out) {
    long long t = blockIdx.x * (long long)blockDim.x + threadIdx.x;
    if (t >= (long long)ET * C) return;
    int e = (int)(t / C);
    int c = (int)(t % C);
    long long s, d;
    if (e < E) { s = ei[e]; d = ei[E + e]; } else { s = d = e - E; }
    int h = c >> hshift;
    float alpha = ex[(size_t)e * Hh + h] / sbuf[d * Hh + h];
    atomicAdd(&out[d * C + c], H[s * C + c] * alpha);
}

// BatchNorm(eval) + ELU, output as f16 (A-matrix for layer-2 GEMM).
__global__ void k_bn_elu(const float* __restrict__ agg, const float* __restrict__ g,
                         const float* __restrict__ bt, const float* __restrict__ mu,
                         const float* __restrict__ var, _Float16* __restrict__ out,
                         long long n) {
    long long t = blockIdx.x * (long long)blockDim.x + threadIdx.x;
    if (t >= n) return;
    int c = (int)(t & 255);
    float v = agg[t];
    v = g[c] * (v - mu[c]) * rsqrtf(var[c] + BN_EPS) + bt[c];
    v = v > 0.f ? v : (__expf(v) - 1.f);
    out[t] = (_Float16)v;
}

// ---------------------------------------------------------------------------
// launch
// ---------------------------------------------------------------------------
static inline long long cdiv(long long a, long long b) { return (a + b - 1) / b; }

extern "C" void kernel_launch(void* const* d_in, const int* in_sizes, int n_in,
                              void* d_out, int out_size, void* d_ws, size_t ws_size,
                              hipStream_t stream) {
    const float*     x      = (const float*)d_in[0];
    const long long* ei     = (const long long*)d_in[1];
    const float*     W1     = (const float*)d_in[2];
    const float*     b_lin1 = (const float*)d_in[3];
    const float*     att1   = (const float*)d_in[4];
    const float*     bias1  = (const float*)d_in[5];
    const float*     bn_g   = (const float*)d_in[6];
    const float*     bn_b   = (const float*)d_in[7];
    const float*     bn_m   = (const float*)d_in[8];
    const float*     bn_v   = (const float*)d_in[9];
    const float*     W2     = (const float*)d_in[10];
    const float*     b_lin2 = (const float*)d_in[11];
    const float*     att2   = (const float*)d_in[12];
    const float*     bias2  = (const float*)d_in[13];
    float* out = (float*)d_out;

    const int Nn = in_sizes[0] / 128;      // 50000
    const int E  = in_sizes[1] / 2;        // 400000
    const int ET = E + Nn;                 // edges incl. self-loops
    const int Mtiles = Nn / 16;            // 3125

    // ---- workspace layout (256B aligned, lifetime-based reuse) ----
    char* ws = (char*)d_ws;
    size_t off = 0;
    auto carve = [&](size_t bytes) { size_t o = off; off += (bytes + 255) & ~(size_t)255; return o; };
    size_t o_xf16   = carve((size_t)Nn * 128 * 2);   // x in f16
    size_t o_bp1    = carve((size_t)128 * 256 * 2);  // packed W1
    size_t o_bp2    = carve((size_t)256 * 64 * 2);   // packed W2
    size_t o_h1     = carve((size_t)Nn * 256 * 4);   // H1 = xW1+b (f32)
    size_t o_lg1    = carve((size_t)ET * 8 * 4);     // logits / exp, layer 1
    size_t o_m1     = carve((size_t)Nn * 8 * 4);     // segment max
    size_t o_s1     = carve((size_t)Nn * 8 * 4);     // segment sum
    size_t o_agg1   = carve((size_t)Nn * 256 * 4);   // aggregated layer-1 output
    size_t o_h1act  = carve((size_t)Nn * 256 * 2);   // BN+ELU output (f16)
    // layer-2 reuse of dead regions:
    size_t o_h2  = o_xf16;                           // Nn*64*4 == Nn*128*2 bytes
    size_t o_lg2 = o_lg1;
    size_t o_m2  = o_m1;
    size_t o_s2  = o_s1;
    (void)ws_size; (void)n_in; (void)out_size;

    _Float16* Xf16  = (_Float16*)(ws + o_xf16);
    _Float16* Bp1   = (_Float16*)(ws + o_bp1);
    _Float16* Bp2   = (_Float16*)(ws + o_bp2);
    float*    H1    = (float*)(ws + o_h1);
    float*    LG1   = (float*)(ws + o_lg1);
    float*    M1    = (float*)(ws + o_m1);
    float*    S1    = (float*)(ws + o_s1);
    float*    AGG1  = (float*)(ws + o_agg1);
    _Float16* H1act = (_Float16*)(ws + o_h1act);
    float*    H2    = (float*)(ws + o_h2);
    float*    LG2   = (float*)(ws + o_lg2);
    float*    M2    = (float*)(ws + o_m2);
    float*    S2    = (float*)(ws + o_s2);

    const int T = 256;
    const float NINF = -__builtin_huge_valf();

    // ---- stage 0: f16 conversion + weight packing ----
    k_cvt_f32_f16<<<(int)cdiv((long long)Nn * 128, T), T, 0, stream>>>(x, Xf16, (long long)Nn * 128);
    k_packw<<<(int)cdiv(128 * 256, T), T, 0, stream>>>(W1, Bp1, 256, 128 * 256);
    k_packw<<<(int)cdiv(256 * 64, T), T, 0, stream>>>(W2, Bp2, 64, 256 * 64);

    // ---- layer 1 GEMM (WMMA): H1 = x @ W1 + b_lin1 ----
    k_gemm_wmma<128, 256><<<(int)cdiv((long long)Mtiles * 16, 8), T, 0, stream>>>(Xf16, Bp1, b_lin1, H1, Mtiles);

    // ---- layer 1 segment softmax + aggregation ----
    k_fill<<<(int)cdiv((long long)Nn * 8, T), T, 0, stream>>>(M1, NINF, (long long)Nn * 8);
    k_fill<<<(int)cdiv((long long)Nn * 8, T), T, 0, stream>>>(S1, 0.f, (long long)Nn * 8);
    k_fill_rowbias<<<(int)cdiv((long long)Nn * 256, T), T, 0, stream>>>(AGG1, bias1, 256, (long long)Nn * 256);

    k_logits1<<<(int)cdiv(ET, 8), T, 0, stream>>>(H1, att1, ei, E, ET, LG1, M1);
    k_expsum<<<(int)cdiv((long long)ET * 8, T), T, 0, stream>>>(LG1, M1, S1, ei, E, ET, 8);
    k_scatter<<<(int)cdiv((long long)ET * 256, T), T, 0, stream>>>(H1, LG1, S1, ei, E, ET, 256, 8, 5, AGG1);

    // ---- BN + ELU -> f16 activations ----
    k_bn_elu<<<(int)cdiv((long long)Nn * 256, T), T, 0, stream>>>(AGG1, bn_g, bn_b, bn_m, bn_v, H1act, (long long)Nn * 256);

    // ---- layer 2 GEMM (WMMA): H2 = h @ W2 + b_lin2 ----
    k_gemm_wmma<256, 64><<<(int)cdiv((long long)Mtiles * 4, 8), T, 0, stream>>>(H1act, Bp2, b_lin2, H2, Mtiles);

    // ---- layer 2 segment softmax + aggregation (into d_out) ----
    k_fill<<<(int)cdiv((long long)Nn, T), T, 0, stream>>>(M2, NINF, (long long)Nn);
    k_fill<<<(int)cdiv((long long)Nn, T), T, 0, stream>>>(S2, 0.f, (long long)Nn);
    k_fill_rowbias<<<(int)cdiv((long long)Nn * 64, T), T, 0, stream>>>(out, bias2, 64, (long long)Nn * 64);

    k_logits2<<<(int)cdiv(ET, 8), T, 0, stream>>>(H2, att2, ei, E, ET, LG2, M2);
    k_expsum<<<(int)cdiv((long long)ET, T), T, 0, stream>>>(LG2, M2, S2, ei, E, ET, 1);
    k_scatter<<<(int)cdiv((long long)ET * 64, T), T, 0, stream>>>(H2, LG2, S2, ei, E, ET, 64, 1, 6, out);
}
